// BasicLSTM_77927886619139
// MI455X (gfx1250) — matrix operand order repaired and checked by reading, not compiled, not run on hardware
//
#include <hip/hip_runtime.h>
#include <hip/hip_bf16.h>
#include <math.h>

typedef __attribute__((ext_vector_type(16))) __bf16 v16bf;
typedef __attribute__((ext_vector_type(8)))  __bf16 v8bf;
typedef __attribute__((ext_vector_type(8)))  float  v8f;
typedef __attribute__((ext_vector_type(4)))  int    v4i;

#define HDIM  256
#define QDIM  32
#define WWIN  27
#define DISL  30
#define BATCH 64
#define TLEN  3000
#define CCH   94
#define NWIN  100
#define EDIM  (CCH*CCH)        // 8836
#define KPAD  8896             // 139 * 64 (zero-padded K)
#define KSTAGES 139            // 64-wide K stages
#define LPITCH  72             // LDS row pitch (bf16): 144B, conflict-free b128
#define MROWS (BATCH*NWIN)     // 6400
#define NGATE 1024             // 4*H

// Async global->LDS path (ASYNCcnt / GLOBAL_LOAD_ASYNC_TO_LDS_B128) if the
// toolchain exposes the builtins; otherwise fall back to register prefetch.
#if defined(__AMDGCN__) && \
    __has_builtin(__builtin_amdgcn_global_load_async_to_lds_b128) && \
    __has_builtin(__builtin_amdgcn_s_wait_asynccnt)
#define USE_ASYNC_LDS 1
#else
#define USE_ASYNC_LDS 0
#endif

#if USE_ASYNC_LDS
typedef __attribute__((address_space(1))) v4i gas_v4i;   // global b128 chunk
typedef __attribute__((address_space(3))) v4i las_v4i;   // LDS b128 chunk
#define ASYNC_CP_B128(dst, src) \
  __builtin_amdgcn_global_load_async_to_lds_b128( \
      (gas_v4i*)(src), (las_v4i*)(dst), 0, 0)
#endif

// ---------------------------------------------------------------------------
// 1) Window correlation -> bf16 X rows (padded to KPAD), row = b*NWIN + n
//    corr = |S_cd| / sqrt(S_cc * S_dd)   (the 1/(W-1) factors cancel).
//    Each thread walks a contiguous run of (row-pair, d) items and caches the
//    two centered rows in registers: 2 FMAs per LDS read instead of 2 reads.
// ---------------------------------------------------------------------------
__global__ __launch_bounds__(256)
void corr_kernel(const float* __restrict__ inp, __bf16* __restrict__ Xb) {
  __shared__ float wS[WWIN*96];
  __shared__ float rstdS[CCH];
  const int bn  = blockIdx.x;            // 0..6399
  const int b   = bn / NWIN;
  const int n   = bn - b*NWIN;
  const int tid = threadIdx.x;

  for (int i = tid; i < WWIN*CCH; i += 256) {
    int w = i / CCH, c = i - w*CCH;
    wS[w*96 + c] = inp[((size_t)b*TLEN + (size_t)n*DISL + w)*CCH + c];
  }
  __syncthreads();

  if (tid < CCH) {
    const int c = tid;
    float m = 0.f;
#pragma unroll
    for (int w = 0; w < WWIN; ++w) m += wS[w*96 + c];
    m *= (1.0f/(float)WWIN);
    float s = 0.f;
#pragma unroll
    for (int w = 0; w < WWIN; ++w) {
      float v = wS[w*96 + c] - m;
      wS[w*96 + c] = v;
      s += v*v;
    }
    rstdS[c] = rsqrtf(s);
  }
  __syncthreads();

  const size_t rowoff = (size_t)bn * KPAD;
  // 47 row-pairs x 94 columns = 4418 items, contiguous chunk per thread
  const int total = (CCH/2) * CCH;            // 4418
  const int chunk = (total + 255) / 256;      // 18
  const int i0 = tid * chunk;
  const int i1 = (i0 + chunk < total) ? (i0 + chunk) : total;
  float a0[WWIN], a1[WWIN];
  int pc = -1;
  for (int idx = i0; idx < i1; ++idx) {
    int p = idx / CCH, d = idx - p*CCH;
    if (p != pc) {
      pc = p;
#pragma unroll
      for (int w = 0; w < WWIN; ++w) {
        a0[w] = wS[w*96 + 2*p];
        a1[w] = wS[w*96 + 2*p + 1];
      }
    }
    float s0 = 0.f, s1 = 0.f;
#pragma unroll
    for (int w = 0; w < WWIN; ++w) {
      float x = wS[w*96 + d];
      s0 += a0[w]*x;
      s1 += a1[w]*x;
    }
    const int c0 = 2*p, c1 = 2*p + 1;
    Xb[rowoff + (size_t)c0*CCH + d] = (__bf16)(fabsf(s0)*rstdS[c0]*rstdS[d]);
    Xb[rowoff + (size_t)c1*CCH + d] = (__bf16)(fabsf(s1)*rstdS[c1]*rstdS[d]);
  }
  for (int e = EDIM + tid; e < KPAD; e += 256) Xb[rowoff + e] = (__bf16)0.f;
}

// ---------------------------------------------------------------------------
// 2) Convert w_ih (1024 x 8836 f32) -> bf16 padded to KPAD columns
// ---------------------------------------------------------------------------
__global__ __launch_bounds__(256)
void cvt_wih_kernel(const float* __restrict__ wih, __bf16* __restrict__ Wb) {
  size_t idx = (size_t)blockIdx.x*256 + threadIdx.x;
  if (idx >= (size_t)NGATE*KPAD) return;
  size_t n = idx / KPAD, k = idx - n*KPAD;
  float v = (k < EDIM) ? wih[n*EDIM + k] : 0.f;
  Wb[idx] = (__bf16)v;
}

// ---------------------------------------------------------------------------
// 3) Convert w_hh -> bf16, bias = b_ih + b_hh
// ---------------------------------------------------------------------------
__global__ __launch_bounds__(256)
void cvt_whh_bias_kernel(const float* __restrict__ whh,
                         const float* __restrict__ bih,
                         const float* __restrict__ bhh,
                         __bf16* __restrict__ Whb,
                         float* __restrict__ bias) {
  int idx = blockIdx.x*256 + threadIdx.x;
  if (idx < NGATE*HDIM) {
    Whb[idx] = (__bf16)whh[idx];
  } else if (idx < NGATE*HDIM + NGATE) {
    int j = idx - NGATE*HDIM;
    bias[j] = bih[j] + bhh[j];
  }
}

// ---------------------------------------------------------------------------
// 4) Big GEMM: xproj[M=6400, N=1024] = Xb[M,KPAD] * Wb[N,KPAD]^T (bf16 WMMA).
//    128x128 tile per WG, 8 waves each 2(M)x4(N) accum tiles, K=64 per LDS
//    stage. Stage s+1 is fetched while stage s computes:
//      - USE_ASYNC_LDS: double-buffered LDS, GLOBAL_LOAD_ASYNC_TO_LDS_B128,
//        s_wait_asynccnt before the stage barrier (no VGPR staging).
//      - fallback: register prefetch + ds_store into a single LDS buffer.
// ---------------------------------------------------------------------------
__global__ __launch_bounds__(256)
void gemm_xproj_kernel(const __bf16* __restrict__ Xb,
                       const __bf16* __restrict__ Wb,
                       float* __restrict__ xproj) {
#if USE_ASYNC_LDS
  __shared__ __bf16 As[2][128*LPITCH];
  __shared__ __bf16 Bs[2][128*LPITCH];
#else
  __shared__ __bf16 As[1][128*LPITCH];
  __shared__ __bf16 Bs[1][128*LPITCH];
#endif
  const int tid  = threadIdx.x;
  const int M0   = blockIdx.y * 128;
  const int N0   = blockIdx.x * 128;
  const int wave = tid >> 5, lane = tid & 31;
  const int wm   = wave & 3;      // rows  wm*32 .. +31  (2 tiles)
  const int wn   = wave >> 2;     // cols  wn*64 .. +63  (4 tiles)
  const int half = lane >> 4;
  const int lm   = lane & 15;
  const int r    = tid >> 1;         // 0..127 staging row
  const int kh   = (tid & 1) * 32;   // 0 / 32 within the 64-wide stage

  const __bf16* gA = Xb + (size_t)(M0 + r)*KPAD + kh;
  const __bf16* gB = Wb + (size_t)(N0 + r)*KPAD + kh;

  v8f acc[2][4] = {};

#if USE_ASYNC_LDS
  // 8 async b128 copies per thread per stage (A: 64B, B: 64B)
  {
    __bf16* dA = &As[0][r*LPITCH + kh];
    __bf16* dB = &Bs[0][r*LPITCH + kh];
    ASYNC_CP_B128(dA,      gA);
    ASYNC_CP_B128(dA + 8,  gA + 8);
    ASYNC_CP_B128(dA + 16, gA + 16);
    ASYNC_CP_B128(dA + 24, gA + 24);
    ASYNC_CP_B128(dB,      gB);
    ASYNC_CP_B128(dB + 8,  gB + 8);
    ASYNC_CP_B128(dB + 16, gB + 16);
    ASYNC_CP_B128(dB + 24, gB + 24);
  }
#else
  // prologue: prefetch stage 0 into registers
  v16bf ra0 = *(const v16bf*)(gA);
  v16bf ra1 = *(const v16bf*)(gA + 16);
  v16bf rb0 = *(const v16bf*)(gB);
  v16bf rb1 = *(const v16bf*)(gB + 16);
#endif

  for (int st = 0; st < KSTAGES; ++st) {
#if USE_ASYNC_LDS
    const int cur = st & 1;
    if (st + 1 < KSTAGES) {           // DMA next stage into the other buffer
      const __bf16* nA = gA + (size_t)(st + 1)*64;
      const __bf16* nB = gB + (size_t)(st + 1)*64;
      __bf16* dA = &As[cur ^ 1][r*LPITCH + kh];
      __bf16* dB = &Bs[cur ^ 1][r*LPITCH + kh];
      ASYNC_CP_B128(dA,      nA);
      ASYNC_CP_B128(dA + 8,  nA + 8);
      ASYNC_CP_B128(dA + 16, nA + 16);
      ASYNC_CP_B128(dA + 24, nA + 24);
      ASYNC_CP_B128(dB,      nB);
      ASYNC_CP_B128(dB + 8,  nB + 8);
      ASYNC_CP_B128(dB + 16, nB + 16);
      ASYNC_CP_B128(dB + 24, nB + 24);
      __builtin_amdgcn_s_wait_asynccnt(8);   // stage `st` copies retired
    } else {
      __builtin_amdgcn_s_wait_asynccnt(0);   // final stage fully resident
    }
    __syncthreads();                         // all waves' copies visible
    const __bf16* Ab = As[cur];
    const __bf16* Bb = Bs[cur];
#else
    *(v16bf*)(&As[0][r*LPITCH + kh])      = ra0;
    *(v16bf*)(&As[0][r*LPITCH + kh + 16]) = ra1;
    *(v16bf*)(&Bs[0][r*LPITCH + kh])      = rb0;
    *(v16bf*)(&Bs[0][r*LPITCH + kh + 16]) = rb1;
    __syncthreads();
    if (st + 1 < KSTAGES) {           // prefetch next stage (overlaps compute)
      const __bf16* nA = gA + (size_t)(st + 1)*64;
      const __bf16* nB = gB + (size_t)(st + 1)*64;
      ra0 = *(const v16bf*)(nA);
      ra1 = *(const v16bf*)(nA + 16);
      rb0 = *(const v16bf*)(nB);
      rb1 = *(const v16bf*)(nB + 16);
    }
    const __bf16* Ab = As[0];
    const __bf16* Bb = Bs[0];
#endif

#pragma unroll
    for (int k2 = 0; k2 < 2; ++k2) {
      // A fragments: lanes 0-15 hold K {0..15}, lanes 16-31 {16..31}
      v16bf af[2];
#pragma unroll
      for (int mt = 0; mt < 2; ++mt) {
        const __bf16* ap = Ab + (wm*32 + mt*16 + lm)*LPITCH + k2*32 + half*16;
        v8bf lo = *(const v8bf*)ap;
        v8bf hi = *(const v8bf*)(ap + 8);
#pragma unroll
        for (int i = 0; i < 8; ++i) { af[mt][i] = lo[i]; af[mt][8+i] = hi[i]; }
      }
#pragma unroll
      for (int nt = 0; nt < 4; ++nt) {
        const __bf16* bp = Bb + (wn*64 + nt*16 + lm)*LPITCH + k2*32 + half*16;
        v16bf bf_ = *(const v16bf*)bp;
#pragma unroll
        for (int mt = 0; mt < 2; ++mt)
          acc[mt][nt] = __builtin_amdgcn_wmma_f32_16x16x32_bf16(
              false, af[mt], false, bf_, (short)0, acc[mt][nt], false, false);
      }
    }
    __syncthreads();
  }

#pragma unroll
  for (int mt = 0; mt < 2; ++mt)
#pragma unroll
    for (int nt = 0; nt < 4; ++nt)
#pragma unroll
      for (int v = 0; v < 8; ++v) {
        int m  = M0 + wm*32 + mt*16 + half*8 + v;   // D: VGPR v -> row v (+8 hi-half)
        int nn = N0 + wn*64 + nt*16 + lm;
        xproj[(size_t)m*NGATE + nn] = acc[mt][nt][v];
      }
}

// ---------------------------------------------------------------------------
// 5) LSTM recurrence. 4 WGs x 16 batches (M = one 16-tile). 16 waves; wave w
//    owns h-cols [16w,16w+16) and computes its i/f/g/o tiles at cols
//    {0,256,512,768}+16w -> all four gates stay in the lane's registers.
// ---------------------------------------------------------------------------
__global__ __launch_bounds__(512)
void lstm_kernel(const float* __restrict__ xproj,
                 const __bf16* __restrict__ Whb,
                 const float* __restrict__ bias,
                 const float* __restrict__ h0,
                 const float* __restrict__ c0,
                 float* __restrict__ outs) {
  __shared__ float  cS[16*HDIM];      // 16 KB
  __shared__ __bf16 hbS[16*264];      // bf16 h, pitch 264 (aligned b128 reads)
  const int tid  = threadIdx.x;
  const int wg   = blockIdx.x;               // batches wg*16 ..
  const int wave = tid >> 5, lane = tid & 31;
  const int half = lane >> 4, lm = lane & 15;

  for (int i = tid; i < 16*HDIM; i += 512) {
    int m = i >> 8, j = i & 255;
    float h = h0[(wg*16 + m)*HDIM + j];
    cS[i] = c0[(wg*16 + m)*HDIM + j];
    hbS[m*264 + j] = (__bf16)h;
  }
  __syncthreads();

  for (int t = 0; t < NWIN; ++t) {
    v8f acc[4] = {};
#pragma unroll
    for (int ks = 0; ks < 8; ++ks) {
      const int k0 = ks * 32;
      const __bf16* ap = hbS + lm*264 + k0 + half*16;
      v8bf lo = *(const v8bf*)ap;
      v8bf hi = *(const v8bf*)(ap + 8);
      v16bf af;
#pragma unroll
      for (int i = 0; i < 8; ++i) { af[i] = lo[i]; af[8+i] = hi[i]; }
#pragma unroll
      for (int nt = 0; nt < 4; ++nt) {
        const int col = nt*HDIM + wave*16 + lm;      // gate nt, h-col wave*16+lm
        v16bf bf_ = *(const v16bf*)(Whb + (size_t)col*HDIM + k0 + half*16);
        acc[nt] = __builtin_amdgcn_wmma_f32_16x16x32_bf16(
            false, af, false, bf_, (short)0, acc[nt], false, false);
      }
    }
    __syncthreads();   // all waves done reading hbS before it is rewritten

    const int j = wave*16 + lm;
#pragma unroll
    for (int v = 0; v < 8; ++v) {
      const int m = half*8 + v;
      const int b = wg*16 + m;
      const size_t xrow = ((size_t)b*NWIN + t)*NGATE;
      float ig = acc[0][v] + xproj[xrow +           j] + bias[          j];
      float fg = acc[1][v] + xproj[xrow +   HDIM +  j] + bias[  HDIM +  j];
      float gg = acc[2][v] + xproj[xrow + 2*HDIM +  j] + bias[2*HDIM +  j];
      float og = acc[3][v] + xproj[xrow + 3*HDIM +  j] + bias[3*HDIM +  j];
      float si = 1.f/(1.f + expf(-ig));
      float sf = 1.f/(1.f + expf(-fg));
      float so = 1.f/(1.f + expf(-og));
      float cn = sf * cS[m*HDIM + j] + si * tanhf(gg);
      float h  = so * tanhf(cn);
      cS[m*HDIM + j] = cn;
      hbS[m*264 + j] = (__bf16)h;
      outs[((size_t)b*NWIN + t)*HDIM + j] = h;
    }
    __syncthreads();   // h visible for next step
  }
}

// ---------------------------------------------------------------------------
// 6) Attention + FC + softmax (tiny). One WG per batch.
//    score = out . (wk @ q) ; z = wv^T @ (p^T out)  (avoids keym/value tensors)
// ---------------------------------------------------------------------------
__global__ __launch_bounds__(128)
void attn_kernel(const float* __restrict__ outs,
                 const float* __restrict__ wq,
                 const float* __restrict__ wk,
                 const float* __restrict__ wv,
                 const float* __restrict__ fc2w,
                 const float* __restrict__ fc2b,
                 float* __restrict__ dout) {
  __shared__ float qS[QDIM];
  __shared__ float uS[HDIM];
  __shared__ float pS[NWIN];
  __shared__ float yS[HDIM];
  __shared__ float zS[QDIM];
  __shared__ float red[1];
  const int b = blockIdx.x, tid = threadIdx.x;
  const float* hT = outs + ((size_t)b*NWIN + (NWIN-1))*HDIM;

  if (tid < QDIM) {
    float s = 0.f;
    for (int j = 0; j < HDIM; ++j) s += hT[j] * wq[j*QDIM + tid];
    qS[tid] = s;
  }
  __syncthreads();
  for (int j = tid; j < HDIM; j += 128) {
    float s = 0.f;
    for (int q = 0; q < QDIM; ++q) s += qS[q] * wk[j*QDIM + q];
    uS[j] = s;
  }
  __syncthreads();
  if (tid < NWIN) {
    const float* orow = outs + ((size_t)b*NWIN + tid)*HDIM;
    float s = 0.f;
    for (int j = 0; j < HDIM; ++j) s += orow[j] * uS[j];
    pS[tid] = s * (1.0f/16.0f);                 // /sqrt(H)
  }
  __syncthreads();
  if (tid == 0) {
    float mx = pS[0];
    for (int s = 1; s < NWIN; ++s) mx = fmaxf(mx, pS[s]);
    float sum = 0.f;
    for (int s = 0; s < NWIN; ++s) { float e = expf(pS[s]-mx); pS[s] = e; sum += e; }
    red[0] = 1.0f/sum;
  }
  __syncthreads();
  const float rs = red[0];
  for (int j = tid; j < HDIM; j += 128) {
    float s = 0.f;
    for (int t2 = 0; t2 < NWIN; ++t2) s += pS[t2] * outs[((size_t)b*NWIN + t2)*HDIM + j];
    yS[j] = s * rs;
  }
  __syncthreads();
  if (tid < QDIM) {
    float s = 0.f;
    for (int j = 0; j < HDIM; ++j) s += yS[j] * wv[j*QDIM + tid];
    zS[tid] = s;
  }
  __syncthreads();
  if (tid == 0) {
    float o0 = fc2b[0], o1 = fc2b[1];
    for (int q = 0; q < QDIM; ++q) { o0 += zS[q]*fc2w[q]; o1 += zS[q]*fc2w[QDIM + q]; }
    o0 = fmaxf(o0, 0.f); o1 = fmaxf(o1, 0.f);
    float mx = fmaxf(o0, o1);
    float e0 = expf(o0 - mx), e1 = expf(o1 - mx);
    float inv = 1.f/(e0 + e1);
    dout[b*2 + 0] = e0*inv;
    dout[b*2 + 1] = e1*inv;
  }
}

// ---------------------------------------------------------------------------
extern "C" void kernel_launch(void* const* d_in, const int* in_sizes, int n_in,
                              void* d_out, int out_size, void* d_ws, size_t ws_size,
                              hipStream_t stream) {
  (void)in_sizes; (void)n_in; (void)out_size; (void)ws_size;
  const float* inputs = (const float*)d_in[0];
  const float* w_ih   = (const float*)d_in[1];
  const float* w_hh   = (const float*)d_in[2];
  const float* b_ih   = (const float*)d_in[3];
  const float* b_hh   = (const float*)d_in[4];
  const float* wq     = (const float*)d_in[5];
  const float* wk     = (const float*)d_in[6];
  const float* wv     = (const float*)d_in[7];
  const float* fc2w   = (const float*)d_in[8];
  const float* fc2b   = (const float*)d_in[9];
  const float* h0     = (const float*)d_in[10];
  const float* c0     = (const float*)d_in[11];
  float* dout = (float*)d_out;

  char* ws = (char*)d_ws;
  size_t off = 0;
  auto carve = [&](size_t bytes) -> char* {
    char* p = ws + off;
    off = (off + bytes + 255) & ~(size_t)255;
    return p;
  };
  __bf16* Xb    = (__bf16*)carve((size_t)MROWS*KPAD*sizeof(__bf16));   // 113.9 MB
  __bf16* Wihb  = (__bf16*)carve((size_t)NGATE*KPAD*sizeof(__bf16));   //  18.2 MB
  __bf16* Whhb  = (__bf16*)carve((size_t)NGATE*HDIM*sizeof(__bf16));   //   0.5 MB
  float*  bias  = (float*) carve((size_t)NGATE*sizeof(float));
  float*  xproj = (float*) carve((size_t)MROWS*NGATE*sizeof(float));   //  26.2 MB
  float*  outs  = (float*) carve((size_t)BATCH*NWIN*HDIM*sizeof(float)); // 6.6 MB

  // weight conversions + correlations (independent; serialized on stream)
  {
    size_t total = (size_t)NGATE*KPAD;
    cvt_wih_kernel<<<(unsigned)((total + 255)/256), 256, 0, stream>>>(w_ih, Wihb);
  }
  {
    int total = NGATE*HDIM + NGATE;
    cvt_whh_bias_kernel<<<(total + 255)/256, 256, 0, stream>>>(w_hh, b_ih, b_hh, Whhb, bias);
  }
  corr_kernel<<<MROWS, 256, 0, stream>>>(inputs, Xb);

  // big input-projection GEMM (bf16 WMMA, f32 accumulate)
  gemm_xproj_kernel<<<dim3(NGATE/128, MROWS/128), 256, 0, stream>>>(Xb, Wihb, xproj);

  // sequential LSTM (batch-parallel, WMMA recurrent GEMM)
  lstm_kernel<<<4, 512, 0, stream>>>(xproj, Whhb, bias, h0, c0, outs);

  // attention + fc + softmax
  attn_kernel<<<BATCH, 128, 0, stream>>>(outs, wq, wk, wv, fc2w, fc2b, dout);
}
